// DecoderBlock_16655883174623
// MI455X (gfx1250) — matrix operand order repaired and checked by reading, not compiled, not run on hardware
//
#include <hip/hip_runtime.h>
#include <hip/hip_bf16.h>

// ---------------------------------------------------------------------------
// Decoder block for MI455X (gfx1250, wave32, WMMA, double-buffered async LDS).
// All GEMMs / attention run as bf16 x bf16 -> f32 via v_wmma_f32_16x16x32_bf16.
// ---------------------------------------------------------------------------

#define BD   8
#define TD   1024
#define SD   1024
#define CD   1024
#define HD_  16
#define HSZ  64
#define MROW (BD * TD)   // 8192

typedef __attribute__((ext_vector_type(16))) __bf16 v16bf;
typedef __attribute__((ext_vector_type(8)))  __bf16 v8bf;
typedef __attribute__((ext_vector_type(8)))  float  v8f;

union V16U { v16bf v; v8bf h[2]; };

// ---------------------------------------------------------------------------
// DPP16 butterfly reduction across 16 lanes (wave32: both halves independent).
// 0xB1 = quad_perm(xor1), 0x4E = quad_perm(xor2),
// 0x141 = row_half_mirror (combines quads), 0x140 = row_mirror (combines 8s).
// ---------------------------------------------------------------------------
template <int CTRL>
__device__ __forceinline__ float dppf(float x) {
  return __int_as_float(__builtin_amdgcn_update_dpp(
      0, __float_as_int(x), CTRL, 0xF, 0xF, true));
}
__device__ __forceinline__ float rmax16(float x) {
  x = fmaxf(x, dppf<0xB1>(x));
  x = fmaxf(x, dppf<0x4E>(x));
  x = fmaxf(x, dppf<0x141>(x));
  x = fmaxf(x, dppf<0x140>(x));
  return x;
}
__device__ __forceinline__ float rsum16(float x) {
  x = x + dppf<0xB1>(x);
  x = x + dppf<0x4E>(x);
  x = x + dppf<0x141>(x);
  x = x + dppf<0x140>(x);
  return x;
}

// ---------------------------------------------------------------------------
// elementwise f32 -> bf16 cast
// ---------------------------------------------------------------------------
__global__ void cast_f32_bf16(const float* __restrict__ in,
                              __bf16* __restrict__ out, int n) {
  int i = blockIdx.x * blockDim.x + threadIdx.x;
  if (i < n) out[i] = (__bf16)in[i];
}

// ---------------------------------------------------------------------------
// fused transpose + cast:  in[R][Cc] f32  ->  out[Cc][R] bf16
// ---------------------------------------------------------------------------
__global__ void transpose_cast(const float* __restrict__ in,
                               __bf16* __restrict__ out, int R, int Cc) {
  __shared__ float t[32][33];
  int bx = blockIdx.x * 32, by = blockIdx.y * 32;
  int tx = threadIdx.x, ty = threadIdx.y;  // 32 x 8
#pragma unroll
  for (int i = 0; i < 32; i += 8)
    t[ty + i][tx] = in[(size_t)(by + ty + i) * Cc + bx + tx];
  __syncthreads();
#pragma unroll
  for (int i = 0; i < 32; i += 8)
    out[(size_t)(bx + ty + i) * R + by + tx] = (__bf16)t[tx][ty + i];
}

// ---------------------------------------------------------------------------
// row LayerNorm: fp32 in (residual), bf16 out. One block per row of C=1024.
// ---------------------------------------------------------------------------
__global__ void layernorm_bf16(const float* __restrict__ x,
                               const float* __restrict__ g,
                               const float* __restrict__ bta,
                               __bf16* __restrict__ out) {
  __shared__ float red[256];
  int row = blockIdx.x;
  int tid = threadIdx.x;
  const float* xr = x + (size_t)row * CD;
  float v4[4];
  float s = 0.f;
#pragma unroll
  for (int i = 0; i < 4; ++i) { v4[i] = xr[tid + i * 256]; s += v4[i]; }
  red[tid] = s; __syncthreads();
  for (int st = 128; st > 0; st >>= 1) { if (tid < st) red[tid] += red[tid + st]; __syncthreads(); }
  float mean = red[0] * (1.0f / CD);
  __syncthreads();
  float s2 = 0.f;
#pragma unroll
  for (int i = 0; i < 4; ++i) { float d = v4[i] - mean; s2 += d * d; }
  red[tid] = s2; __syncthreads();
  for (int st = 128; st > 0; st >>= 1) { if (tid < st) red[tid] += red[tid + st]; __syncthreads(); }
  float rstd = rsqrtf(red[0] * (1.0f / CD) + 1e-5f);
#pragma unroll
  for (int i = 0; i < 4; ++i) {
    int c = tid + i * 256;
    out[(size_t)row * CD + c] = (__bf16)((v4[i] - mean) * rstd * g[c] + bta[c]);
  }
}

// ---------------------------------------------------------------------------
// async global -> LDS: moves 32B (two b128, INST_OFFSET applies to both the
// global and the LDS address per ISA 15.18.3). Tracked by ASYNCcnt, in order.
// ---------------------------------------------------------------------------
__device__ __forceinline__ void async_copy32(const __bf16* gsrc, __bf16* ldst) {
  unsigned laddr = (unsigned)(uintptr_t)ldst;                 // low 32 bits = LDS offset
  unsigned long long ga = (unsigned long long)(uintptr_t)gsrc;
  asm volatile("global_load_async_to_lds_b128 %0, %1, off"
               :: "v"(laddr), "v"(ga) : "memory");
  asm volatile("global_load_async_to_lds_b128 %0, %1, off offset:16"
               :: "v"(laddr), "v"(ga) : "memory");
}
template <int N>
__device__ __forceinline__ void wait_async() {
  asm volatile("s_wait_asynccnt %0" :: "i"(N) : "memory");
}

// ---------------------------------------------------------------------------
// Tiled WMMA GEMM:  D[M,N] = A[M,K] (bf16, row major) x Wt[N,K] (bf16, W^T)
// block = 256 threads = 8 waves; block tile 128x128; wave tile 64x32.
// Double-buffered async staging: issue tile i+1, s_wait_asynccnt 4 drains
// only tile i (async ops complete in order), WMMAs overlap the new loads.
// MODE 0: store bf16          MODE 1: bias+relu -> bf16
// MODE 2: resid += acc+bias   MODE 3: store bf16 transposed as [b][c][s]
// ---------------------------------------------------------------------------
template <int MODE>
__global__ void gemm_bf16_wmma(const __bf16* __restrict__ A,
                               const __bf16* __restrict__ Wt,
                               int Kdim, int Ndim,
                               const float* __restrict__ bias,
                               float* __restrict__ resid,
                               __bf16* __restrict__ outb) {
  __shared__ __attribute__((aligned(32))) __bf16 sA[2][128][32];
  __shared__ __attribute__((aligned(32))) __bf16 sW[2][128][32];   // [n][k]

  int tid = threadIdx.x;
  int wave = tid >> 5, lane = tid & 31, half = lane >> 4, lm = lane & 15;
  int wm = wave >> 2, wn = wave & 3;
  int mBlk = blockIdx.y * 128, nBlk = blockIdx.x * 128;

  v8f acc[4][2];
#pragma unroll
  for (int mi = 0; mi < 4; ++mi)
#pragma unroll
    for (int ni = 0; ni < 2; ++ni)
#pragma unroll
      for (int e = 0; e < 8; ++e) acc[mi][ni][e] = 0.f;

  int arow = tid >> 1, acol = (tid & 1) * 16;   // both tiles are 128x32
  const __bf16* gA0 = A  + (size_t)(mBlk + arow) * Kdim + acol;
  const __bf16* gW0 = Wt + (size_t)(nBlk + arow) * Kdim + acol;

  // prologue: stage K-tile 0 into buffer 0  (ASYNCcnt = 4)
  async_copy32(gA0, &sA[0][arow][acol]);
  async_copy32(gW0, &sW[0][arow][acol]);

  int it = 0;
  for (int k0 = 0; k0 < Kdim; k0 += 32, ++it) {
    int cur = it & 1, nxt = cur ^ 1;
    if (k0 + 32 < Kdim) {
      // stage next tile while we compute this one
      async_copy32(gA0 + k0 + 32, &sA[nxt][arow][acol]);
      async_copy32(gW0 + k0 + 32, &sW[nxt][arow][acol]);
      __builtin_prefetch(gA0 + k0 + 64, 0, 0);   // global_prefetch_b8, 2 tiles ahead
      __builtin_prefetch(gW0 + k0 + 64, 0, 0);
      wait_async<4>();   // drain only the 4 older ops (current tile)
    } else {
      wait_async<0>();
    }
    __syncthreads();

    V16U af[4], bw[2];
#pragma unroll
    for (int mi = 0; mi < 4; ++mi) {
      const __bf16* ap = &sA[cur][wm * 64 + mi * 16 + lm][half * 16];
      af[mi].h[0] = *(const v8bf*)(ap);
      af[mi].h[1] = *(const v8bf*)(ap + 8);
    }
#pragma unroll
    for (int ni = 0; ni < 2; ++ni)
      bw[ni].v = *(const v16bf*)&sW[cur][wn * 32 + ni * 16 + lm][half * 16];
#pragma unroll
    for (int mi = 0; mi < 4; ++mi)
#pragma unroll
      for (int ni = 0; ni < 2; ++ni)
        acc[mi][ni] = __builtin_amdgcn_wmma_f32_16x16x32_bf16(
            false, af[mi].v, false, bw[ni].v, (short)0, acc[mi][ni], false, false);
    __syncthreads();   // all waves done reading 'cur' before it is re-staged
  }

#pragma unroll
  for (int mi = 0; mi < 4; ++mi)
#pragma unroll
    for (int ni = 0; ni < 2; ++ni)
#pragma unroll
      for (int vv = 0; vv < 8; ++vv) {
        int row = mBlk + wm * 64 + mi * 16 + vv + 8 * half;  // C layout: M = v + 8*half
        int col = nBlk + wn * 32 + ni * 16 + lm;             // N = lane%16
        float val = acc[mi][ni][vv];
        if (MODE == 0) {
          outb[(size_t)row * Ndim + col] = (__bf16)val;
        } else if (MODE == 1) {
          outb[(size_t)row * Ndim + col] = (__bf16)fmaxf(val + bias[col], 0.f);
        } else if (MODE == 2) {
          resid[(size_t)row * Ndim + col] += val + bias[col];
        } else {  // MODE 3: v written as [b][c][s] for fast attention B-frags
          int bb = row >> 10, ss = row & 1023;
          outb[((size_t)bb * CD + col) * SD + ss] = (__bf16)val;
        }
      }
}

// ---------------------------------------------------------------------------
// Flash attention: grid (T/128, H, B), 256 threads = 8 waves, 16 q-rows/wave.
// q/k bf16 [B, T|S, C] head-major channels; v bf16 transposed [B][C][S].
// ---------------------------------------------------------------------------
__global__ void attn_flash(const __bf16* __restrict__ q,
                           const __bf16* __restrict__ k,
                           const __bf16* __restrict__ vt,
                           __bf16* __restrict__ o,
                           int causal) {
  __shared__ __attribute__((aligned(32))) __bf16 pbuf[8][16][32];  // per-wave P staging

  int tid = threadIdx.x;
  int wave = tid >> 5, lane = tid & 31, half = lane >> 4, lm = lane & 15;
  int b = blockIdx.z, h = blockIdx.y;
  int qBase = blockIdx.x * 128 + wave * 16;

  // Q A-fragments (16x64); softmax scale folded into scores post-WMMA
  V16U aq0, aq1;
  {
    const __bf16* qrow = q + ((size_t)(b * TD + qBase + lm) * CD) + h * HSZ;
    aq0.h[0] = *(const v8bf*)(qrow + half * 16);
    aq0.h[1] = *(const v8bf*)(qrow + 8 + half * 16);
    aq1.h[0] = *(const v8bf*)(qrow + 32 + half * 16);
    aq1.h[1] = *(const v8bf*)(qrow + 40 + half * 16);
  }
  const __bf16* vbase = vt + ((size_t)b * CD + h * HSZ) * SD;

  v8f oacc[4];
  float mr[8], lr[8];
#pragma unroll
  for (int vv = 0; vv < 8; ++vv) { mr[vv] = -3.0e38f; lr[vv] = 0.f; }
#pragma unroll
  for (int nt = 0; nt < 4; ++nt)
#pragma unroll
    for (int vv = 0; vv < 8; ++vv) oacc[nt][vv] = 0.f;

  int jEnd = causal ? (((qBase + 16 + 31) & ~31) < SD ? ((qBase + 16 + 31) & ~31) : SD) : SD;

  for (int j = 0; j < jEnd; j += 32) {
    // K^T B-fragments: two 16-key subtiles x two 32-wide d halves
    v16bf bk[2][2];
#pragma unroll
    for (int t2 = 0; t2 < 2; ++t2) {
      const __bf16* krow = k + ((size_t)(b * SD + j + t2 * 16 + lm) * CD) + h * HSZ + half * 16;
      bk[t2][0] = *(const v16bf*)(const void*)(krow);
      bk[t2][1] = *(const v16bf*)(const void*)(krow + 32);
    }
    // V B-fragments issued early so they overlap QK^T + softmax + P staging
    v16bf bv[4];
#pragma unroll
    for (int nt = 0; nt < 4; ++nt)
      bv[nt] = *(const v16bf*)(const void*)(vbase + (size_t)(nt * 16 + lm) * SD + j + 16 * half);

    v8f c[2];
#pragma unroll
    for (int t2 = 0; t2 < 2; ++t2) {
      v8f cc;
#pragma unroll
      for (int e = 0; e < 8; ++e) cc[e] = 0.f;
      cc = __builtin_amdgcn_wmma_f32_16x16x32_bf16(false, aq0.v, false, bk[t2][0], (short)0, cc, false, false);
      cc = __builtin_amdgcn_wmma_f32_16x16x32_bf16(false, aq1.v, false, bk[t2][1], (short)0, cc, false, false);
#pragma unroll
      for (int e = 0; e < 8; ++e) cc[e] *= 0.125f;  // 1/sqrt(64)
      c[t2] = cc;
    }

    if (causal) {
#pragma unroll
      for (int t2 = 0; t2 < 2; ++t2)
#pragma unroll
        for (int vv = 0; vv < 8; ++vv)
          if (j + t2 * 16 + lm > qBase + vv + 8 * half) c[t2][vv] = -3.0e38f;
    }

    // online softmax; row reductions via DPP16 butterflies (no LDS, no waits)
    float alpha[8];
#pragma unroll
    for (int vv = 0; vv < 8; ++vv) {
      float mx = rmax16(fmaxf(c[0][vv], c[1][vv]));
      float mnew = fmaxf(mr[vv], mx);
      alpha[vv] = __expf(mr[vv] - mnew);
      float p0 = __expf(c[0][vv] - mnew);
      float p1 = __expf(c[1][vv] - mnew);
      c[0][vv] = p0; c[1][vv] = p1;
      float sum = rsum16(p0 + p1);
      lr[vv] = lr[vv] * alpha[vv] + sum;
      mr[vv] = mnew;
    }
#pragma unroll
    for (int nt = 0; nt < 4; ++nt)
#pragma unroll
      for (int vv = 0; vv < 8; ++vv) oacc[nt][vv] *= alpha[vv];

    // relayout P: C layout -> A fragment through per-wave LDS tile
#pragma unroll
    for (int t2 = 0; t2 < 2; ++t2)
#pragma unroll
      for (int vv = 0; vv < 8; ++vv)
        pbuf[wave][vv + 8 * half][t2 * 16 + lm] = (__bf16)c[t2][vv];
    V16U pa;
    {
      const __bf16* pr = &pbuf[wave][lm][0];
      pa.h[0] = *(const v8bf*)(pr + half * 16);
      pa.h[1] = *(const v8bf*)(pr + 8 + half * 16);
    }

    // P (16x32) x V (32x64): 4 back-to-back WMMAs, fragments already resident
#pragma unroll
    for (int nt = 0; nt < 4; ++nt)
      oacc[nt] = __builtin_amdgcn_wmma_f32_16x16x32_bf16(
          false, pa.v, false, bv[nt], (short)0, oacc[nt], false, false);
  }

  float inv[8];
#pragma unroll
  for (int vv = 0; vv < 8; ++vv) inv[vv] = 1.0f / lr[vv];
#pragma unroll
  for (int nt = 0; nt < 4; ++nt)
#pragma unroll
    for (int vv = 0; vv < 8; ++vv) {
      int row = qBase + vv + 8 * half;
      o[((size_t)(b * TD + row) * CD) + h * HSZ + nt * 16 + lm] =
          (__bf16)(oacc[nt][vv] * inv[vv]);
    }
}

// ---------------------------------------------------------------------------
// host orchestration
// ---------------------------------------------------------------------------
extern "C" void kernel_launch(void* const* d_in, const int* in_sizes, int n_in,
                              void* d_out, int out_size, void* d_ws, size_t ws_size,
                              hipStream_t stream) {
  const float* x    = (const float*)d_in[0];
  const float* enc  = (const float*)d_in[1];
  const float* sa_wq = (const float*)d_in[2];
  const float* sa_wk = (const float*)d_in[3];
  const float* sa_wv = (const float*)d_in[4];
  const float* sa_pw = (const float*)d_in[5];
  const float* sa_pb = (const float*)d_in[6];
  const float* ca_wq = (const float*)d_in[7];
  const float* ca_wk = (const float*)d_in[8];
  const float* ca_wv = (const float*)d_in[9];
  const float* ca_pw = (const float*)d_in[10];
  const float* ca_pb = (const float*)d_in[11];
  const float* ff_w1 = (const float*)d_in[12];
  const float* ff_b1 = (const float*)d_in[13];
  const float* ff_w2 = (const float*)d_in[14];
  const float* ff_b2 = (const float*)d_in[15];
  const float* ln1_g = (const float*)d_in[16];
  const float* ln1_b = (const float*)d_in[17];
  const float* ln2_g = (const float*)d_in[18];
  const float* ln2_b = (const float*)d_in[19];
  const float* ln3_g = (const float*)d_in[20];
  const float* ln3_b = (const float*)d_in[21];
  float* out = (float*)d_out;

  const size_t CC = (size_t)1024 * 1024;
  __bf16* ws = (__bf16*)d_ws;
  __bf16 *w_saq = ws,        *w_sak = ws + CC,   *w_sav = ws + 2*CC, *w_sap = ws + 3*CC;
  __bf16 *w_caq = ws + 4*CC, *w_cak = ws + 5*CC, *w_cav = ws + 6*CC, *w_cap = ws + 7*CC;
  __bf16 *w_ff1 = ws + 8*CC, *w_ff2 = ws + 12*CC;
  __bf16 *xbf   = ws + 16*CC, *lnbf = ws + 24*CC, *encbf = ws + 32*CC;
  __bf16 *qbf   = ws + 40*CC, *kbf  = ws + 48*CC, *vbf   = ws + 56*CC;
  __bf16 *abf   = ws + 64*CC, *hbf  = ws + 72*CC;  // hbf: 32*CC elems

  // residual lives in d_out (fp32)
  hipMemcpyAsync(out, x, (size_t)MROW * CD * sizeof(float),
                 hipMemcpyDeviceToDevice, stream);

  auto cast = [&](const float* src, __bf16* dst, int n) {
    cast_f32_bf16<<<(n + 255) / 256, 256, 0, stream>>>(src, dst, n);
  };
  // weights: transpose+cast to bf16 [N][K]
  auto tcast = [&](const float* src, __bf16* dst, int R, int Cc) {
    transpose_cast<<<dim3(Cc / 32, R / 32), dim3(32, 8), 0, stream>>>(src, dst, R, Cc);
  };
  cast(x, xbf, MROW * CD);
  cast(enc, encbf, BD * SD * CD);
  tcast(sa_wq, w_saq, CD, CD); tcast(sa_wk, w_sak, CD, CD);
  tcast(sa_wv, w_sav, CD, CD); tcast(sa_pw, w_sap, CD, CD);
  tcast(ca_wq, w_caq, CD, CD); tcast(ca_wk, w_cak, CD, CD);
  tcast(ca_wv, w_cav, CD, CD); tcast(ca_pw, w_cap, CD, CD);
  tcast(ff_w1, w_ff1, CD, 4 * CD); tcast(ff_w2, w_ff2, 4 * CD, CD);

  auto gemm = [&](const __bf16* A, const __bf16* Wt, int K, int N,
                  const float* bias, float* resid, __bf16* ob, int mode) {
    dim3 grid(N / 128, MROW / 128);
    switch (mode) {
      case 0:  gemm_bf16_wmma<0><<<grid, 256, 0, stream>>>(A, Wt, K, N, bias, resid, ob); break;
      case 1:  gemm_bf16_wmma<1><<<grid, 256, 0, stream>>>(A, Wt, K, N, bias, resid, ob); break;
      case 2:  gemm_bf16_wmma<2><<<grid, 256, 0, stream>>>(A, Wt, K, N, bias, resid, ob); break;
      default: gemm_bf16_wmma<3><<<grid, 256, 0, stream>>>(A, Wt, K, N, bias, resid, ob); break;
    }
  };
  dim3 agrid(TD / 128, HD_, BD);

  // ---- self-attention: q from ln1(x), k/v from raw x, causal ----
  layernorm_bf16<<<MROW, 256, 0, stream>>>(out, ln1_g, ln1_b, lnbf);
  gemm(lnbf, w_saq, CD, CD, nullptr, nullptr, qbf, 0);
  gemm(xbf,  w_sak, CD, CD, nullptr, nullptr, kbf, 0);
  gemm(xbf,  w_sav, CD, CD, nullptr, nullptr, vbf, 3);
  attn_flash<<<agrid, 256, 0, stream>>>(qbf, kbf, vbf, abf, 1);
  gemm(abf, w_sap, CD, CD, sa_pb, out, nullptr, 2);

  // ---- cross-attention: q from ln2(x), k/v from encoder, no mask ----
  layernorm_bf16<<<MROW, 256, 0, stream>>>(out, ln2_g, ln2_b, lnbf);
  gemm(lnbf,  w_caq, CD, CD, nullptr, nullptr, qbf, 0);
  gemm(encbf, w_cak, CD, CD, nullptr, nullptr, kbf, 0);
  gemm(encbf, w_cav, CD, CD, nullptr, nullptr, vbf, 3);
  attn_flash<<<agrid, 256, 0, stream>>>(qbf, kbf, vbf, abf, 0);
  gemm(abf, w_cap, CD, CD, ca_pb, out, nullptr, 2);

  // ---- feed-forward ----
  layernorm_bf16<<<MROW, 256, 0, stream>>>(out, ln3_g, ln3_b, lnbf);
  gemm(lnbf, w_ff1, CD, 4 * CD, ff_b1, nullptr, hbf, 1);
  gemm(hbf,  w_ff2, 4 * CD, CD, ff_b2, out, nullptr, 2);
}
